// DeepseekV4MoE_36472862277791
// MI455X (gfx1250) — matrix operand order repaired and checked by reading, not compiled, not run on hardware
//
#include <hip/hip_runtime.h>
#include <hip/hip_bf16.h>
#include <math.h>

#define T_TOK 4096
#define DIM   2048
#define IDIM  1024
#define NEXP  32
#define TOPK  4
#define ALPHA 7.0f
#define MTOK  64          // tokens per workgroup (4 M-tiles of 16)
#define ACT_STRIDE 2064   // 1024 f16 = 2048B + 16B pad (bank-conflict break, keeps 16B align)
#define XA_STRIDE  80     // 32 f16 = 64B + 16B pad

typedef __attribute__((ext_vector_type(16))) _Float16  v16h;
typedef __attribute__((ext_vector_type(2)))  _Float16  v2h;
typedef __attribute__((ext_vector_type(8)))  float     v8f;
typedef __attribute__((ext_vector_type(8)))  unsigned  v8u;

// ---- packed f32->f16 conversion: ONE VALU op (v_cvt_pk_rtz_f16_f32) per pair ----
static __device__ inline unsigned cvtpk(float a, float b) {
#if __has_builtin(__builtin_amdgcn_cvt_pkrtz)
    return __builtin_bit_cast(unsigned, __builtin_amdgcn_cvt_pkrtz(a, b));
#else
    v2h r; r[0] = (_Float16)a; r[1] = (_Float16)b;
    return __builtin_bit_cast(unsigned, r);
#endif
}

static __device__ inline v8f wmma16(v16h a, v16h b, v8f c) {
    return __builtin_amdgcn_wmma_f32_16x16x32_f16(false, a, false, b, (short)0, c, false, false);
}

// B-matrix (32x16 f16): lane<16 -> col N=lane, K = kb2..kb2+15 across v0..7 (pairs)
// caller passes kb2 = kblock + (lane>>4)*16 ; row = weight row W[n][*] (row-major, K contiguous)
static __device__ inline v16h load_b_f32(const float* __restrict__ row, int kb2) {
    const float4* p = (const float4*)(row + kb2);
    float4 f0 = p[0], f1 = p[1], f2 = p[2], f3 = p[3];
    v8u u;
    u[0] = cvtpk(f0.x, f0.y); u[1] = cvtpk(f0.z, f0.w);
    u[2] = cvtpk(f1.x, f1.y); u[3] = cvtpk(f1.z, f1.w);
    u[4] = cvtpk(f2.x, f2.y); u[5] = cvtpk(f2.z, f2.w);
    u[6] = cvtpk(f3.x, f3.y); u[7] = cvtpk(f3.z, f3.w);
    return __builtin_bit_cast(v16h, u);
}

// A-matrix (16x32 f16) from LDS: lane<16 -> row M=lane, K = kb..kb+15 (v0..7)
// p points at (row, k-chunk base + hi*16) in bytes; second half at +16B.
static __device__ inline v16h load_a_lds(const char* p) {
    uint4 a0 = *(const uint4*)(p);
    uint4 a1 = *(const uint4*)(p + 16);
    v8u u; u[0]=a0.x; u[1]=a0.y; u[2]=a0.z; u[3]=a0.w;
           u[4]=a1.x; u[5]=a1.y; u[6]=a1.z; u[7]=a1.w;
    return __builtin_bit_cast(v16h, u);
}

// ---------------- Kernel 1: router ----------------
__global__ __launch_bounds__(256) void router_kernel(
    const float* __restrict__ X, const float* __restrict__ Rw,
    const float* __restrict__ bias, int* __restrict__ sel_e, float* __restrict__ sel_w)
{
    int t    = blockIdx.x * 8 + (threadIdx.x >> 5);
    int lane = threadIdx.x & 31;               // lane == expert index
    const float* x = X  + (size_t)t    * DIM;
    const float* w = Rw + (size_t)lane * DIM;
    float acc = 0.f;
    for (int d = 0; d < DIM; d += 4) {
        float4 xv = *(const float4*)(x + d);
        float4 wv = *(const float4*)(w + d);
        acc += xv.x*wv.x + xv.y*wv.y + xv.z*wv.z + xv.w*wv.w;
    }
    float sp     = (acc > 20.f) ? acc : log1pf(__expf(acc));   // stable softplus
    float score  = sqrtf(sp);
    float biased = score + bias[lane];

    float sum = 0.f; int wl[TOPK]; float wsc[TOPK];
    #pragma unroll
    for (int r = 0; r < TOPK; ++r) {
        float bv = biased; int bl = lane;
        #pragma unroll
        for (int off = 16; off > 0; off >>= 1) {
            float ov = __shfl_xor(bv, off, 32);
            int   ol = __shfl_xor(bl, off, 32);
            if (ov > bv || (ov == bv && ol < bl)) { bv = ov; bl = ol; }
        }
        float ws = __shfl(score, bl, 32);      // raw (unbiased) score of winner
        wl[r] = bl; wsc[r] = ws; sum += ws;
        if (lane == bl) biased = -3.4e38f;
    }
    if (lane == 0) {
        float inv = 1.f / (sum + 1e-20f);
        #pragma unroll
        for (int r = 0; r < TOPK; ++r) {
            sel_e[t*TOPK + r] = wl[r];
            sel_w[t*TOPK + r] = wsc[r] * inv;
        }
    }
}

// ---------------- Kernel 2: deterministic per-expert token lists ----------------
__global__ __launch_bounds__(128) void build_lists_kernel(
    const int* __restrict__ sel_e, const float* __restrict__ sel_w,
    int* __restrict__ cnt, int* __restrict__ tok_idx, float* __restrict__ tok_w)
{
    int e = blockIdx.x, tid = threadIdx.x;     // 128 threads x 32 tokens
    __shared__ int s_cnt[128];
    int base = tid * 32, c = 0;
    for (int t = base; t < base + 32; ++t)
        for (int j = 0; j < TOPK; ++j)
            if (sel_e[t*TOPK + j] == e) c++;
    s_cnt[tid] = c;
    __syncthreads();
    for (int off = 1; off < 128; off <<= 1) {  // inclusive scan
        int v = (tid >= off) ? s_cnt[tid - off] : 0;
        __syncthreads();
        s_cnt[tid] += v;
        __syncthreads();
    }
    int pos = s_cnt[tid] - c;                  // exclusive offset (token-ordered)
    for (int t = base; t < base + 32; ++t)
        for (int j = 0; j < TOPK; ++j)
            if (sel_e[t*TOPK + j] == e) {
                tok_idx[e*T_TOK + pos] = t;
                tok_w  [e*T_TOK + pos] = sel_w[t*TOPK + j];
                pos++;
            }
    if (tid == 127) cnt[e] = s_cnt[127];
}

// ---------------- Kernel 3: fused expert MLP (WMMA f16, fp32 accumulate) ----------------
// grid: (64 tiles of 64 tokens, 33 experts)  y==32 -> shared expert (all tokens, w=1)
// Each WG: 64 gathered tokens (4 M-tiles), 8 waves. B tiles amortized over 4 M-tiles.
__global__ __launch_bounds__(256) void moe_mlp_kernel(
    const float* __restrict__ X,
    const float* __restrict__ gate_w, const float* __restrict__ up_w,
    const float* __restrict__ down_w,
    const float* __restrict__ sgw, const float* __restrict__ suw,
    const float* __restrict__ sdw,
    const int* __restrict__ cnt, const int* __restrict__ tok_idx,
    const float* __restrict__ tok_w, float* __restrict__ out)
{
    __shared__ char  s_act[MTOK * ACT_STRIDE];   // 64 x 1024 f16 (padded rows) = 132KB
    __shared__ char  s_xa [MTOK * XA_STRIDE];    // 64 x 32 f16 K-chunk of X (padded) = 5KB
    __shared__ int   s_tok[MTOK];
    __shared__ float s_w  [MTOK];

    const int e = blockIdx.y, tile = blockIdx.x, tid = threadIdx.x;
    const int lane = tid & 31, wave = tid >> 5;
    const float *Wg, *Wu, *Wd;

    if (e < NEXP) {
        const int c = cnt[e], start = tile * MTOK;
        if (start >= c) return;                // uniform across block
        if (tid < MTOK) {
            int idx = start + tid;
            if (idx < c) { s_tok[tid] = tok_idx[e*T_TOK + idx]; s_w[tid] = tok_w[e*T_TOK + idx]; }
            else         { s_tok[tid] = 0;                      s_w[tid] = 0.f; }
        }
        Wg = gate_w + (size_t)e * IDIM * DIM;
        Wu = up_w   + (size_t)e * IDIM * DIM;
        Wd = down_w + (size_t)e * DIM  * IDIM;
    } else {                                   // shared expert: all tokens, weight 1
        if (tid < MTOK) { s_tok[tid] = tile * MTOK + tid; s_w[tid] = 1.f; }
        Wg = sgw; Wu = suw; Wd = sdw;
    }
    __syncthreads();

    const int m  = lane & 15;                  // row/col within a 16-tile
    const int hi = lane >> 4;
    // staging role: thread -> (row, 8-float segment) of the 64x32 X chunk
    const int sr   = tid >> 2;                 // 0..63
    const int sseg = (tid & 3) * 8;            // 0,8,16,24
    const float* s_row_ptr = X + (size_t)s_tok[sr] * DIM + sseg;

    // ---- Stage A: act[64 x 1024] = clamped-SwiGLU(X@Wg^T, X@Wu^T), 4 N-passes ----
    for (int pass = 0; pass < 4; ++pass) {
        v8f accG[2][4], accU[2][4];
        #pragma unroll
        for (int t = 0; t < 2; ++t)
            #pragma unroll
            for (int mt = 0; mt < 4; ++mt) { accG[t][mt] = (v8f){}; accU[t][mt] = (v8f){}; }

        for (int kk = 0; kk < DIM; kk += 32) {
            // cooperative stage of X[64][kk..kk+32) -> f16 LDS
            {
                const float4* sp = (const float4*)(s_row_ptr + kk);
                float4 a = sp[0], b = sp[1];
                uint4 o;
                o.x = cvtpk(a.x, a.y); o.y = cvtpk(a.z, a.w);
                o.z = cvtpk(b.x, b.y); o.w = cvtpk(b.z, b.w);
                *(uint4*)(s_xa + sr * XA_STRIDE + sseg * 2) = o;
            }
            __syncthreads();

            v16h A[4];
            #pragma unroll
            for (int mt = 0; mt < 4; ++mt)
                A[mt] = load_a_lds(s_xa + (mt * 16 + m) * XA_STRIDE + hi * 32);

            #pragma unroll
            for (int t = 0; t < 2; ++t) {
                int ntile = pass * 16 + wave * 2 + t;
                const float* gRow = Wg + (size_t)(ntile * 16 + m) * DIM;
                const float* uRow = Wu + (size_t)(ntile * 16 + m) * DIM;
                __builtin_prefetch(gRow + kk + 32, 0, 3);
                v16h Bg = load_b_f32(gRow, kk + hi * 16);
                #pragma unroll
                for (int mt = 0; mt < 4; ++mt) accG[t][mt] = wmma16(A[mt], Bg, accG[t][mt]);
                __builtin_prefetch(uRow + kk + 32, 0, 3);
                v16h Bu = load_b_f32(uRow, kk + hi * 16);
                #pragma unroll
                for (int mt = 0; mt < 4; ++mt) accU[t][mt] = wmma16(A[mt], Bu, accU[t][mt]);
            }
            __syncthreads();   // protect s_xa before next restage
        }

        // activation: clamp +/-7, silu(g)*u -> f16 act in LDS
        #pragma unroll
        for (int t = 0; t < 2; ++t) {
            int n = (pass * 16 + wave * 2 + t) * 16 + m;   // act column
            #pragma unroll
            for (int mt = 0; mt < 4; ++mt) {
                #pragma unroll
                for (int i = 0; i < 8; ++i) {
                    float g = fminf(fmaxf(accG[t][mt][i], -ALPHA), ALPHA);
                    float u = fminf(fmaxf(accU[t][mt][i], -ALPHA), ALPHA);
                    float a = (g / (1.f + __expf(-g))) * u;
                    int row = mt * 16 + i + hi * 8;
                    _Float16 h = (_Float16)a;
                    *(unsigned short*)(s_act + row * ACT_STRIDE + n * 2) =
                        __builtin_bit_cast(unsigned short, h);
                }
            }
        }
    }
    __syncthreads();   // act fully written

    // ---- Stage B: out[64 x 2048] = act @ Wd^T, 8 N-passes (Wd read exactly once/WG) ----
    for (int pass = 0; pass < 8; ++pass) {
        v8f acc[2][4];
        #pragma unroll
        for (int t = 0; t < 2; ++t)
            #pragma unroll
            for (int mt = 0; mt < 4; ++mt) acc[t][mt] = (v8f){};

        for (int kk = 0; kk < IDIM; kk += 32) {
            v16h A[4];
            #pragma unroll
            for (int mt = 0; mt < 4; ++mt)
                A[mt] = load_a_lds(s_act + (mt * 16 + m) * ACT_STRIDE + (kk + hi * 16) * 2);
            #pragma unroll
            for (int t = 0; t < 2; ++t) {
                int ntile = pass * 16 + wave * 2 + t;
                const float* dRow = Wd + (size_t)(ntile * 16 + m) * IDIM;
                __builtin_prefetch(dRow + kk + 32, 0, 3);
                v16h B = load_b_f32(dRow, kk + hi * 16);
                #pragma unroll
                for (int mt = 0; mt < 4; ++mt) acc[t][mt] = wmma16(A[mt], B, acc[t][mt]);
            }
        }
        #pragma unroll
        for (int t = 0; t < 2; ++t) {
            int n = (pass * 16 + wave * 2 + t) * 16 + m;   // output column
            #pragma unroll
            for (int mt = 0; mt < 4; ++mt) {
                #pragma unroll
                for (int i = 0; i < 8; ++i) {
                    int row = mt * 16 + i + hi * 8;
                    float v = acc[t][mt][i] * s_w[row];
                    atomicAdd(out + (size_t)s_tok[row] * DIM + n, v);
                }
            }
        }
    }
}

// ---------------- launch ----------------
extern "C" void kernel_launch(void* const* d_in, const int* in_sizes, int n_in,
                              void* d_out, int out_size, void* d_ws, size_t ws_size,
                              hipStream_t stream) {
    (void)in_sizes; (void)n_in; (void)ws_size;
    const float* hidden   = (const float*)d_in[0];
    const float* router_w = (const float*)d_in[1];
    const float* exp_bias = (const float*)d_in[2];
    const float* gate_w   = (const float*)d_in[3];
    const float* up_w     = (const float*)d_in[4];
    const float* down_w   = (const float*)d_in[5];
    const float* sgw      = (const float*)d_in[6];
    const float* suw      = (const float*)d_in[7];
    const float* sdw      = (const float*)d_in[8];
    float* out = (float*)d_out;

    char* ws = (char*)d_ws;
    int*   sel_e   = (int*)  (ws);                         // T*4 ints
    float* sel_w   = (float*)(ws + 1*65536);               // T*4 floats
    int*   cnt     = (int*)  (ws + 2*65536);               // 32 ints (pad 256B)
    int*   tok_idx = (int*)  (ws + 2*65536 + 256);         // E*T ints
    float* tok_w   = (float*)(ws + 2*65536 + 256 + 524288);// E*T floats

    hipMemsetAsync(d_out, 0, (size_t)out_size * sizeof(float), stream);

    router_kernel<<<T_TOK / 8, 256, 0, stream>>>(hidden, router_w, exp_bias, sel_e, sel_w);
    build_lists_kernel<<<NEXP, 128, 0, stream>>>(sel_e, sel_w, cnt, tok_idx, tok_w);
    moe_mlp_kernel<<<dim3(T_TOK / MTOK, NEXP + 1), 256, 0, stream>>>(
        hidden, gate_w, up_w, down_w, sgw, suw, sdw, cnt, tok_idx, tok_w, out);
}
